// StackRNN_54245436948936
// MI455X (gfx1250) — matrix operand order repaired and checked by reading, not compiled, not run on hardware
//
#include <hip/hip_runtime.h>
#include <hip/hip_bf16.h>

// Problem constants (from reference)
#define S_   128
#define B_   128
#define IN_  512
#define H_   512
#define SW_  64
#define SD_  32
#define FOURH_ 2048
#define M_   (S_ * B_)          // 16384 rows in the big GEMMs

typedef __attribute__((ext_vector_type(16))) _Float16 v16h;
typedef __attribute__((ext_vector_type(8)))  _Float16 f16x8;
typedef __attribute__((ext_vector_type(8)))  float    v8f;

__device__ __forceinline__ v16h join8(f16x8 lo, f16x8 hi) {
    return __builtin_shufflevector(lo, hi, 0,1,2,3,4,5,6,7,8,9,10,11,12,13,14,15);
}

__device__ __forceinline__ float wredsum(float v) {
    #pragma unroll
    for (int o = 16; o > 0; o >>= 1) v += __shfl_xor(v, o, 32);
    return v;
}

__device__ __forceinline__ float sigmoidf_(float x) {
    return 1.0f / (1.0f + __expf(-x));
}

__device__ __forceinline__ void wait_async0() {
#if defined(__has_builtin)
#if __has_builtin(__builtin_amdgcn_s_wait_asynccnt)
    __builtin_amdgcn_s_wait_asynccnt(0);
#else
    asm volatile("s_wait_asynccnt 0x0" ::: "memory");
#endif
#else
    asm volatile("s_wait_asynccnt 0x0" ::: "memory");
#endif
}

// ---------------------------------------------------------------------------
// 1) Stack recurrence. h0 == 0 so controls = softmax(bc) (batch-invariant
//    scalars), s_in = tanh(bs). The stack is identical for every batch elem;
//    compute the per-timestep stack tops tops[t][k] once. 64 threads, each
//    owns one stack column (SD=32 registers).
// ---------------------------------------------------------------------------
__global__ void prep_stack_kernel(const float* __restrict__ bc0, const float* __restrict__ bs0,
                                  const float* __restrict__ bc1, const float* __restrict__ bs1,
                                  float* __restrict__ tops0, float* __restrict__ tops1)
{
    const int k = threadIdx.x;           // 0..63
    for (int l = 0; l < 2; ++l) {
        const float* bc = l ? bc1 : bc0;
        const float* bs = l ? bs1 : bs0;
        float* tops     = l ? tops1 : tops0;

        float b0 = bc[0], b1 = bc[1], b2 = bc[2];
        float mx = fmaxf(b0, fmaxf(b1, b2));
        float e0 = __expf(b0 - mx), e1 = __expf(b1 - mx), e2 = __expf(b2 - mx);
        float inv = 1.0f / (e0 + e1 + e2);
        float a_push = e0 * inv, a_pop = e1 * inv, a_noop = e2 * inv;

        float s_in = tanhf(bs[k]);

        float col[SD_];
        #pragma unroll
        for (int d = 0; d < SD_; ++d) col[d] = 0.0f;

        for (int t = 0; t < S_; ++t) {
            float up = s_in;             // stack_up row 0 is s_in
            #pragma unroll
            for (int d = 0; d < SD_; ++d) {
                float oldv = col[d];
                float down = (d < SD_ - 1) ? col[d + 1] : 0.0f;
                col[d] = a_noop * oldv + a_push * up + a_pop * down;
                up = oldv;               // old stack[d] feeds stack_up[d+1]
            }
            tops[t * SW_ + k] = col[0];  // top AFTER update, as in reference
        }
    }
}

// ---------------------------------------------------------------------------
// 2) f32 -> f16 conversion with row remap (drops the SW tail of Wih rows).
// ---------------------------------------------------------------------------
__global__ void cvt_f16_kernel(const float* __restrict__ src, _Float16* __restrict__ dst,
                               int rows, int srcStride, int cols)
{
    size_t n = (size_t)rows * cols;
    for (size_t i = (size_t)blockIdx.x * blockDim.x + threadIdx.x; i < n;
         i += (size_t)gridDim.x * blockDim.x) {
        size_t r = i / cols, c = i - r * cols;
        dst[i] = (_Float16)src[r * (size_t)srcStride + c];
    }
}

// ---------------------------------------------------------------------------
// 3) Per-timestep gate bias: bias[t][n] = tops[t] . Wih[n, IN:IN+SW] + bih[n] + bhh[n]
// ---------------------------------------------------------------------------
__global__ void biasmat_kernel(const float* __restrict__ tops, const float* __restrict__ Wih,
                               const float* __restrict__ bih, const float* __restrict__ bhh,
                               float* __restrict__ out)
{
    int idx = blockIdx.x * blockDim.x + threadIdx.x;     // S_*FOURH_
    if (idx >= S_ * FOURH_) return;
    int t = idx >> 11;               // / 2048
    int n = idx & (FOURH_ - 1);
    const float* w  = Wih + (size_t)n * (IN_ + SW_) + IN_;
    const float* tp = tops + t * SW_;
    float s = bih[n] + bhh[n];
    #pragma unroll 8
    for (int k = 0; k < SW_; ++k) s = fmaf(tp[k], w[k], s);
    out[idx] = s;
}

// ---------------------------------------------------------------------------
// 4) Main WMMA GEMM + fused gate activation.
//    gates[m, n] = Xh[m, :] . Wh[n, :]  (n only over gate groups i(0), g(2), o(3))
//    c = sigmoid(i)*tanh(g)  (f gate dead: c0 == 0),  h = sigmoid(o)*tanh(c)
//    Block: 192 threads = 6 waves = 3 gates x 2 row-halves; wave tile 64x32.
//    A tile (128x32 f16, 8KB) is staged once per K-step into LDS with
//    GLOBAL_LOAD_ASYNC_TO_LDS_B128 (ASYNCcnt), removing the 3x duplicated
//    per-wave A gathers; waves then read fragments with ds_load_b128.
// ---------------------------------------------------------------------------
__global__ __launch_bounds__(192) void gemm_gates_kernel(
    const _Float16* __restrict__ Xh,       // [M_, 512] f16 row-major
    const _Float16* __restrict__ Wh,       // [2048, 512] f16 row-major
    const float*    __restrict__ biasmat,  // [S_, 2048]
    float* __restrict__ hbuf, float* __restrict__ cbuf)  // [M_, 512]
{
    __shared__ _Float16 ldsA[128 * 32];    // 8KB A-tile stage (one K-step)
    __shared__ float    gacc[3][128][33];  // 48KB gate staging (+1 pad)

    const int tid   = threadIdx.x;
    const int wave  = tid >> 5;            // 0..5
    const int lane  = tid & 31;
    const int slot  = wave % 3;            // 0->i, 1->g, 2->o
    const int gateIdx = (slot == 0) ? 0 : (slot == 1) ? 2 : 3;
    const int mhalf = wave / 3;            // 0..1
    const int l15   = lane & 15;
    const int lhi   = lane >> 4;

    const int blockM = blockIdx.y * 128;
    const int hTile  = blockIdx.x * 32;

    const v8f vzero = {0.f,0.f,0.f,0.f,0.f,0.f,0.f,0.f};
    v8f acc[4][2];
    #pragma unroll
    for (int mi = 0; mi < 4; ++mi)
        #pragma unroll
        for (int ni = 0; ni < 2; ++ni) acc[mi][ni] = vzero;

    for (int kk = 0; kk < 512; kk += 32) {
        // --- Stage A tile: thread r copies row (blockM+r), 64B = 4 x b128.
        // INST_OFFSET adds to BOTH global and LDS addresses (ISA 08 §4.4).
        if (tid < 128) {
            const _Float16* grow = Xh + (size_t)(blockM + tid) * 512 + kk;
            unsigned lla = (unsigned)(uintptr_t)(ldsA + tid * 32);
            unsigned long long ga = (unsigned long long)(uintptr_t)grow;
            asm volatile(
                "global_load_async_to_lds_b128 %0, %1, off\n\t"
                "global_load_async_to_lds_b128 %0, %1, off offset:16\n\t"
                "global_load_async_to_lds_b128 %0, %1, off offset:32\n\t"
                "global_load_async_to_lds_b128 %0, %1, off offset:48"
                :: "v"(lla), "v"(ga) : "memory");
            __builtin_prefetch(grow + 32, 0, 3);   // next K chunk
        }

        // --- B fragments direct from global (per-gate weights, L2 resident).
        // B[k][n] = Wh[n][k]; lane&15 -> column n, lane>>4 picks K in {0..7,16..23} or {8..15,24..31}.
        v16h b[2];
        #pragma unroll
        for (int ni = 0; ni < 2; ++ni) {
            const _Float16* p = Wh + (size_t)(gateIdx * 512 + hTile + ni * 16 + l15) * 512
                                   + kk + lhi * 8;
            f16x8 lo = *(const f16x8*)p;
            f16x8 hi = *(const f16x8*)(p + 16);
            b[ni] = join8(lo, hi);
            __builtin_prefetch(p + 32, 0, 3);      // next K chunk (speculative)
        }

        wait_async0();
        __syncthreads();

        // --- A fragments from LDS. ISA 16-bit A layout: lane<16 -> row=lane,
        // K in {0..7,16..23}; lane>=16 -> row=lane-16, K in {8..15,24..31}.
        v16h a[4];
        #pragma unroll
        for (int mi = 0; mi < 4; ++mi) {
            const _Float16* lrow = ldsA + (size_t)(mhalf * 64 + mi * 16 + l15) * 32 + lhi * 8;
            f16x8 lo = *(const f16x8*)lrow;
            f16x8 hi = *(const f16x8*)(lrow + 16);
            a[mi] = join8(lo, hi);
        }

        #pragma unroll
        for (int mi = 0; mi < 4; ++mi)
            #pragma unroll
            for (int ni = 0; ni < 2; ++ni)
                acc[mi][ni] = __builtin_amdgcn_wmma_f32_16x16x32_f16(
                    false, a[mi], false, b[ni], (short)0, acc[mi][ni], false, false);

        __syncthreads();   // A stage reused next iteration
    }

    // C layout: lanes 0-15 -> N=lane, M=v; lanes 16-31 -> N=lane-16, M=8+v.
    #pragma unroll
    for (int mi = 0; mi < 4; ++mi)
        #pragma unroll
        for (int ni = 0; ni < 2; ++ni)
            #pragma unroll
            for (int v = 0; v < 8; ++v)
                gacc[slot][mhalf * 64 + mi * 16 + lhi * 8 + v][ni * 16 + l15] = acc[mi][ni][v];

    __syncthreads();

    // Fused gate activation; f gate dead since c0 == 0.
    for (int e = tid; e < 128 * 32; e += 192) {
        int row = e >> 5, col = e & 31;
        int m = blockM + row;
        int t = m >> 7;                       // m / B_
        int hc = hTile + col;
        const float* bm = biasmat + (size_t)t * FOURH_;
        float gi = gacc[0][row][col] + bm[hc];
        float gg = gacc[1][row][col] + bm[1024 + hc];
        float go = gacc[2][row][col] + bm[1536 + hc];
        float c  = sigmoidf_(gi) * tanhf(gg);
        float h  = sigmoidf_(go) * tanhf(c);
        size_t o = (size_t)m * H_ + hc;
        hbuf[o] = h;
        cbuf[o] = c;
    }
}

// ---------------------------------------------------------------------------
// 5) LayerNorm stage. One wave per row of H=512 (16 elems/lane).
//    mode 0: y1 = LN(h; gh,bh); x1 = relu(LN(y1; gx,bx)) -> f16; at t==S-1
//            write y1 -> hn_last and LN(c; gh,bh) -> cn_last.
//    mode 1: y1 = LN(h; gh,bh) -> full d_out; t==S-1 slices as above.
// ---------------------------------------------------------------------------
__global__ __launch_bounds__(256) void ln_kernel(
    const float* __restrict__ hbuf, const float* __restrict__ cbuf,
    const float* __restrict__ gh, const float* __restrict__ bh,
    const float* __restrict__ gx, const float* __restrict__ bx,
    _Float16* __restrict__ x1h, float* __restrict__ out_full,
    float* __restrict__ hn_last, float* __restrict__ cn_last, int mode)
{
    const int wave = threadIdx.x >> 5;
    const int lane = threadIdx.x & 31;
    const int m = blockIdx.x * 8 + wave;
    const int t = m >> 7;
    const int bi = m & (B_ - 1);

    const float* row = hbuf + (size_t)m * H_;
    float v[16];
    float s = 0.f;
    #pragma unroll
    for (int i = 0; i < 16; ++i) { v[i] = row[lane + i * 32]; s += v[i]; }
    s = wredsum(s);
    float mean = s * (1.0f / H_);
    float q = 0.f;
    #pragma unroll
    for (int i = 0; i < 16; ++i) { v[i] -= mean; q += v[i] * v[i]; }
    q = wredsum(q);
    float inv = rsqrtf(q * (1.0f / H_) + 1e-5f);

    float y[16];
    #pragma unroll
    for (int i = 0; i < 16; ++i) {
        int c = lane + i * 32;
        y[i] = v[i] * inv * gh[c] + bh[c];
    }

    if (mode == 0) {
        float s2 = 0.f;
        #pragma unroll
        for (int i = 0; i < 16; ++i) s2 += y[i];
        s2 = wredsum(s2);
        float m2 = s2 * (1.0f / H_);
        float q2 = 0.f;
        #pragma unroll
        for (int i = 0; i < 16; ++i) { float d = y[i] - m2; q2 += d * d; }
        q2 = wredsum(q2);
        float inv2 = rsqrtf(q2 * (1.0f / H_) + 1e-5f);
        #pragma unroll
        for (int i = 0; i < 16; ++i) {
            int c = lane + i * 32;
            float z = (y[i] - m2) * inv2 * gx[c] + bx[c];
            x1h[(size_t)m * H_ + c] = (_Float16)fmaxf(z, 0.0f);
        }
    } else {
        #pragma unroll
        for (int i = 0; i < 16; ++i)
            out_full[(size_t)m * H_ + lane + i * 32] = y[i];
    }

    if (t == S_ - 1) {
        #pragma unroll
        for (int i = 0; i < 16; ++i)
            hn_last[(size_t)bi * H_ + lane + i * 32] = y[i];

        const float* crow = cbuf + (size_t)m * H_;
        float cv[16];
        float cs = 0.f;
        #pragma unroll
        for (int i = 0; i < 16; ++i) { cv[i] = crow[lane + i * 32]; cs += cv[i]; }
        cs = wredsum(cs);
        float cm = cs * (1.0f / H_);
        float cq = 0.f;
        #pragma unroll
        for (int i = 0; i < 16; ++i) { cv[i] -= cm; cq += cv[i] * cv[i]; }
        cq = wredsum(cq);
        float cinv = rsqrtf(cq * (1.0f / H_) + 1e-5f);
        #pragma unroll
        for (int i = 0; i < 16; ++i) {
            int c = lane + i * 32;
            cn_last[(size_t)bi * H_ + c] = cv[i] * cinv * gh[c] + bh[c];
        }
    }
}

// ---------------------------------------------------------------------------
// Host launcher
// ---------------------------------------------------------------------------
extern "C" void kernel_launch(void* const* d_in, const int* in_sizes, int n_in,
                              void* d_out, int out_size, void* d_ws, size_t ws_size,
                              hipStream_t stream)
{
    (void)in_sizes; (void)n_in; (void)out_size; (void)ws_size;

    const float* x    = (const float*)d_in[0];
    const float* Wih0 = (const float*)d_in[1];
    const float* bih0 = (const float*)d_in[3];
    const float* bhh0 = (const float*)d_in[4];
    const float* bc0  = (const float*)d_in[6];
    const float* bs0  = (const float*)d_in[8];
    const float* Wih1 = (const float*)d_in[9];
    const float* bih1 = (const float*)d_in[11];
    const float* bhh1 = (const float*)d_in[12];
    const float* bc1  = (const float*)d_in[14];
    const float* bs1  = (const float*)d_in[16];
    const float* gh   = (const float*)d_in[17];
    const float* bh   = (const float*)d_in[18];
    const float* gx   = (const float*)d_in[19];
    const float* bx   = (const float*)d_in[20];

    // Workspace carve-up (256B aligned regions)
    char* p = (char*)d_ws;
    auto carve = [&](size_t bytes) -> char* {
        char* r = p;
        p += (bytes + 255) & ~(size_t)255;
        return r;
    };
    float*    tops0 = (float*)   carve((size_t)S_ * SW_ * 4);
    float*    tops1 = (float*)   carve((size_t)S_ * SW_ * 4);
    float*    bias0 = (float*)   carve((size_t)S_ * FOURH_ * 4);
    float*    bias1 = (float*)   carve((size_t)S_ * FOURH_ * 4);
    float*    hbuf  = (float*)   carve((size_t)M_ * H_ * 4);
    float*    cbuf  = (float*)   carve((size_t)M_ * H_ * 4);
    _Float16* Xh    = (_Float16*)carve((size_t)M_ * IN_ * 2);
    _Float16* X1h   = (_Float16*)carve((size_t)M_ * H_ * 2);
    _Float16* W0h   = (_Float16*)carve((size_t)FOURH_ * IN_ * 2);
    _Float16* W1h   = (_Float16*)carve((size_t)FOURH_ * H_ * 2);

    // d_out layout: out (S,B,H) | h_n (2,B,H) | c_n (2,B,H)
    float* out      = (float*)d_out;
    float* hn0_last = out + (size_t)S_ * B_ * H_;
    float* hn1_last = hn0_last + (size_t)B_ * H_;
    float* cn0_last = hn1_last + (size_t)B_ * H_;
    float* cn1_last = cn0_last + (size_t)B_ * H_;

    // 1) Stack tops (batch-invariant)
    prep_stack_kernel<<<1, 64, 0, stream>>>(bc0, bs0, bc1, bs1, tops0, tops1);

    // 2) f16 conversions
    cvt_f16_kernel<<<4096, 256, 0, stream>>>(x, Xh, M_, IN_, IN_);
    cvt_f16_kernel<<<1024, 256, 0, stream>>>(Wih0, W0h, FOURH_, IN_ + SW_, IN_);
    cvt_f16_kernel<<<1024, 256, 0, stream>>>(Wih1, W1h, FOURH_, H_ + SW_, H_);

    // 3) Per-timestep gate biases (stack-top contribution + bih + bhh)
    biasmat_kernel<<<(S_ * FOURH_) / 256, 256, 0, stream>>>(tops0, Wih0, bih0, bhh0, bias0);
    biasmat_kernel<<<(S_ * FOURH_) / 256, 256, 0, stream>>>(tops1, Wih1, bih1, bhh1, bias1);

    dim3 ggrid(H_ / 32, M_ / 128);

    // 4) Layer 0: GEMM + gates -> h0, c0
    gemm_gates_kernel<<<ggrid, 192, 0, stream>>>(Xh, W0h, bias0, hbuf, cbuf);

    // 5) LN stage 0: x1 = relu(LN(LN(h0))), last-step slices
    ln_kernel<<<M_ / 8, 256, 0, stream>>>(hbuf, cbuf, gh, bh, gx, bx,
                                          X1h, out, hn0_last, cn0_last, 0);

    // 6) Layer 1: GEMM + gates -> h1, c1 (reuse buffers)
    gemm_gates_kernel<<<ggrid, 192, 0, stream>>>(X1h, W1h, bias1, hbuf, cbuf);

    // 7) LN stage 1: out = LN(h1), last-step slices
    ln_kernel<<<M_ / 8, 256, 0, stream>>>(hbuf, cbuf, gh, bh, gx, bx,
                                          X1h, out, hn1_last, cn1_last, 1);
}
